// FlashMLAAttention_38620345925801
// MI455X (gfx1250) — matrix operand order repaired and checked by reading, not compiled, not run on hardware
//
#include <hip/hip_runtime.h>
#include <hip/hip_bf16.h>

typedef __bf16 bf16;
typedef __attribute__((ext_vector_type(16))) __bf16 v16bf;
typedef __attribute__((ext_vector_type(8)))  float  v8f;

// ---------------------------------------------------------------------------
// Helpers
// ---------------------------------------------------------------------------
struct B32x8 { uint4 lo, hi; };          // 32 bytes

static __device__ __forceinline__ v16bf load_frag2(const bf16* p0, const bf16* p1) {
  B32x8 u;
  u.lo = *(const uint4*)p0;
  u.hi = *(const uint4*)p1;
  return __builtin_bit_cast(v16bf, u);
}

static __device__ __forceinline__ v8f wmma_bf16(v16bf a, v16bf b, v8f c) {
  // D = A(16x32) * B(32x16) + C, f32 accumulate
  return __builtin_amdgcn_wmma_f32_16x16x32_bf16(
      /*neg_a=*/false, a, /*neg_b=*/false, b,
      /*c_mod=*/(short)0, c, /*reuse_a=*/false, /*reuse_b=*/false);
}

// --- CDNA5 async global->LDS staging (ASYNCcnt path), with safe fallback ----
#if __has_builtin(__builtin_amdgcn_global_load_async_to_lds_b128)
#define HAVE_ASYNC_LDS 1
typedef int v4i_vs __attribute__((vector_size(16)));
typedef __attribute__((address_space(1))) v4i_vs as1_v4i;  // global
typedef __attribute__((address_space(3))) v4i_vs as3_v4i;  // LDS
#endif

static __device__ __forceinline__ void stage16(void* lds, const void* g) {
#ifdef HAVE_ASYNC_LDS
  __builtin_amdgcn_global_load_async_to_lds_b128(
      (as1_v4i*)(unsigned long long)g,
      (as3_v4i*)(unsigned int)(unsigned long long)lds, 0, 0);
#else
  *(uint4*)lds = *(const uint4*)g;
#endif
}

static __device__ __forceinline__ void stage_wait() {
#ifdef HAVE_ASYNC_LDS
#if __has_builtin(__builtin_amdgcn_s_wait_asynccnt)
  __builtin_amdgcn_s_wait_asynccnt(0);
#else
  asm volatile("s_wait_asynccnt 0x0" ::: "memory");
#endif
#endif
}

// ---------------------------------------------------------------------------
// Cast / transpose kernels (f32 -> bf16)
// ---------------------------------------------------------------------------
__global__ __launch_bounds__(256) void cast_bf16_kernel(
    const float* __restrict__ s, bf16* __restrict__ d, long n) {
  long i = (long)blockIdx.x * blockDim.x + threadIdx.x;
  long stride = (long)gridDim.x * blockDim.x;
  for (; i < n; i += stride) d[i] = (bf16)s[i];
}

// d[c*R + r] = (bf16) s[r*C + c];  R, C multiples of 32
__global__ __launch_bounds__(256) void transpose_bf16_kernel(
    const float* __restrict__ s, bf16* __restrict__ d, int R, int C) {
  __shared__ bf16 tile[32][33];
  int c0 = blockIdx.x * 32, r0 = blockIdx.y * 32;
  int tx = threadIdx.x & 31, ty = threadIdx.x >> 5;  // 8 rows per pass
  for (int i = 0; i < 32; i += 8)
    tile[ty + i][tx] = (bf16)s[(long)(r0 + ty + i) * C + (c0 + tx)];
  __syncthreads();
  for (int i = 0; i < 32; i += 8)
    d[(long)(c0 + ty + i) * R + (r0 + tx)] = tile[tx][ty + i];
}

// ---------------------------------------------------------------------------
// Generic bf16 WMMA GEMM:  C[M,N] = A[M,K] * B[K,N]
// A row-major bf16 (lda), B row-major bf16 (ldb), C bf16 or f32 (template).
// Optional head batching along grid.z with element offsets a_hs/b_hs/c_hs.
// Block 128 = 4 waves; tile 64x128, BK=32; waves 2x2, each 32x64 -> 8 WMMA/step.
// ---------------------------------------------------------------------------
#define GEMM_BM 64
#define GEMM_BN 128
#define GEMM_BK 32
#define LDS_STR 48   // halfs; 96B row stride (16B multiple)

template <bool F32OUT>
__global__ __launch_bounds__(128) void gemm_bf16_kernel(
    const bf16* __restrict__ A, long lda, long a_hs,
    const bf16* __restrict__ B, long ldb, long b_hs,
    void* __restrict__ Cp, long ldc, long c_hs,
    int M, int N, int K) {
  __shared__ bf16 As[GEMM_BM][LDS_STR];   // [m][k]
  __shared__ bf16 Bs[GEMM_BN][LDS_STR];   // [n][k]  (transposed stage)
  const int hz = blockIdx.z;
  A += (long)hz * a_hs;
  B += (long)hz * b_hs;
  const long coff = (long)hz * c_hs;
  const int n0 = blockIdx.x * GEMM_BN;
  const int m0 = blockIdx.y * GEMM_BM;
  const int t = threadIdx.x;
  const int wave = t >> 5, lane = t & 31;
  const int half = lane >> 4, lr = lane & 15;
  const int wm = (wave >> 1) * 32;        // wave covers 32 (M) x 64 (N)
  const int wn = (wave & 1) * 64;

  v8f acc[2][4] = {};

  for (int k0 = 0; k0 < K; k0 += GEMM_BK) {
    // stage A tile 64x32 via async global->LDS (256 chunks of 16B)
    #pragma unroll
    for (int i = 0; i < 2; ++i) {
      int ch = i * 128 + t;
      int r = ch >> 2, c = (ch & 3) * 8;
      const bf16* gp = &A[(long)(m0 + r) * lda + k0 + c];
      stage16(&As[r][c], gp);
      if (k0 + GEMM_BK < K) __builtin_prefetch(gp + GEMM_BK, 0, 1);
    }
    // stage B tile 32x128 transposed into Bs[n][k] (scatter, 512 chunks)
    #pragma unroll
    for (int i = 0; i < 4; ++i) {
      int ch = i * 128 + t;
      int kk = ch >> 4, nn = (ch & 15) * 8;
      uint4 v = *(const uint4*)&B[(long)(k0 + kk) * ldb + n0 + nn];
      const bf16* pv = (const bf16*)&v;
      #pragma unroll
      for (int j = 0; j < 8; ++j) Bs[nn + j][kk] = pv[j];
    }
    stage_wait();
    __syncthreads();

    v16bf af[2], bfv[4];
    #pragma unroll
    for (int i = 0; i < 2; ++i) {
      // A 16x32 layout: lanes<16 hold K{0-7,16-23}, lanes>=16 hold K{8-15,24-31}
      const bf16* p = &As[wm + i * 16 + lr][half * 8];
      af[i] = load_frag2(p, p + 16);
    }
    #pragma unroll
    for (int j = 0; j < 4; ++j) {
      // B 32x16 layout: lane = column, lanes<16 K{0-7,16-23}, lanes>=16 K{8-15,24-31}
      const bf16* p = &Bs[wn + j * 16 + lr][half * 8];
      bfv[j] = load_frag2(p, p + 16);
    }
    #pragma unroll
    for (int i = 0; i < 2; ++i)
      #pragma unroll
      for (int j = 0; j < 4; ++j)
        acc[i][j] = wmma_bf16(af[i], bfv[j], acc[i][j]);
    __syncthreads();
  }

  // C layout: VGPR r -> rows r (lanes 0-15) / r+8 (lanes 16-31), col = lane&15
  float* __restrict__ Cf = (float*)Cp;
  bf16*  __restrict__ Cb = (bf16*)Cp;
  #pragma unroll
  for (int i = 0; i < 2; ++i)
    #pragma unroll
    for (int j = 0; j < 4; ++j)
      #pragma unroll
      for (int r = 0; r < 8; ++r) {
        long row = m0 + wm + i * 16 + r + half * 8;
        long col = n0 + wn + j * 16 + lr;
        float v = acc[i][j][r];
        if constexpr (F32OUT) Cf[coff + row * ldc + col] = v;
        else                  Cb[coff + row * ldc + col] = (bf16)v;
      }
}

// ---------------------------------------------------------------------------
// Flash MLA attention (causal, online softmax), bf16 WMMA.
//   Qabs : [B*T, 16, 512]  per-head latent queries
//   Cl   : [B*T, 512]      latent KV (keys)
//   Vh   : [B*T, 2048]     expanded V (head h at column h*128)
//   O    : [B*T, 2048]     attention output (bf16)
// Grid: (B*16, T/64); block 128 = 4 waves; wave w owns query rows q0+16w..+15.
// K tile staged in LDS (async) once per key tile; V tile staged transposed.
// ---------------------------------------------------------------------------
#define ATT_QT 64
#define ATT_ST 64
#define KS_STR 520   // halfs; 1040B = 65*16B (aligned, conflict-breaking)
#define VT_STR 80    // halfs; 160B
#define PW_STR 80

__global__ __launch_bounds__(128) void mla_attn_kernel(
    const bf16* __restrict__ Qabs, const bf16* __restrict__ Cl,
    const bf16* __restrict__ Vh, bf16* __restrict__ O, int T) {
  __shared__ bf16 Ks[64][KS_STR];         // K tile row-major: [key][latent]
  __shared__ bf16 Vt[128][VT_STR];        // V tile transposed: [n][key]
  __shared__ bf16 Pw[4][16][PW_STR];      // per-wave probabilities (A layout)

  const int bh = blockIdx.x;
  const int b = bh >> 4, h = bh & 15;
  const int q0 = blockIdx.y * ATT_QT;
  const int t = threadIdx.x, wave = t >> 5, lane = t & 31;
  const int half = lane >> 4, lr = lane & 15;
  const long rowQ = ((long)(b * T + q0 + wave * 16 + lr) * 16 + h) * 512;
  const float SCALE = 0.08838834764831845f;  // 128^-0.5

  float mrow[8], lrow[8];
  v8f o[8] = {};
  #pragma unroll
  for (int r = 0; r < 8; ++r) { mrow[r] = -1e30f; lrow[r] = 0.f; }

  const int nkt = q0 / ATT_ST + 1;        // causal: only tiles with s0 <= q0
  for (int kt = 0; kt < nkt; ++kt) {
    const int s0 = kt * ATT_ST;

    // stage K tile (64 keys x 512 latents) via async copy: 4096 chunks of 16B
    for (int i = 0; i < 32; ++i) {
      int ch = i * 128 + t;
      int key = ch >> 6, cc = (ch & 63) * 8;
      stage16(&Ks[key][cc], &Cl[(long)(b * T + s0 + key) * 512 + cc]);
    }
    // stage V tile (64 keys x 128 dims) transposed into LDS
    #pragma unroll
    for (int i = 0; i < 8; ++i) {
      int ch = i * 128 + t;               // 1024 chunks of 8 halfs
      int key = ch >> 4, nn = (ch & 15) * 8;
      uint4 v = *(const uint4*)&Vh[(long)(b * T + s0 + key) * 2048 + h * 128 + nn];
      const bf16* pv = (const bf16*)&v;
      #pragma unroll
      for (int j = 0; j < 8; ++j) Vt[nn + j][key] = pv[j];
    }
    stage_wait();
    __syncthreads();

    // --- S = Q K^T : 16 (rows) x 64 (keys), contraction over 512 latents ---
    v8f s[4] = {};
    for (int kb = 0; kb < 512; kb += 32) {
      const bf16* qp = Qabs + rowQ + kb;
      v16bf aq = load_frag2(qp + half * 8, qp + 16 + half * 8);
      #pragma unroll
      for (int nt = 0; nt < 4; ++nt) {
        const bf16* kp = &Ks[nt * 16 + lr][kb + half * 8];
        v16bf bk = load_frag2(kp, kp + 16);
        s[nt] = wmma_bf16(aq, bk, s[nt]);
      }
    }

    // --- scale, causal mask, online softmax update ---
    #pragma unroll
    for (int r = 0; r < 8; ++r) {
      const int rowg = q0 + wave * 16 + r + half * 8;
      float mnew = mrow[r];
      float sv[4];
      #pragma unroll
      for (int nt = 0; nt < 4; ++nt) {
        float v = s[nt][r] * SCALE;
        if (s0 + nt * 16 + lr > rowg) v = -1e30f;
        sv[nt] = v;
        mnew = fmaxf(mnew, v);
      }
      #pragma unroll
      for (int d = 1; d < 16; d <<= 1)      // stays inside 16-lane half
        mnew = fmaxf(mnew, __shfl_xor(mnew, d, 32));
      const float alpha = __expf(mrow[r] - mnew);
      float psum = 0.f;
      #pragma unroll
      for (int nt = 0; nt < 4; ++nt) {
        float p = __expf(sv[nt] - mnew);
        psum += p;
        Pw[wave][r + half * 8][nt * 16 + lr] = (bf16)p;
      }
      #pragma unroll
      for (int d = 1; d < 16; d <<= 1)
        psum += __shfl_xor(psum, d, 32);
      lrow[r] = lrow[r] * alpha + psum;
      mrow[r] = mnew;
      #pragma unroll
      for (int nt2 = 0; nt2 < 8; ++nt2) o[nt2][r] *= alpha;
    }

    // intra-wave LDS RAW: ensure P stores land before A-fragment reloads
    asm volatile("s_wait_dscnt 0" ::: "memory");

    // --- O += P (16x64) * V (64x128) ---
    #pragma unroll
    for (int kc = 0; kc < 64; kc += 32) {
      const bf16* pp = &Pw[wave][lr][kc + half * 8];
      v16bf ap = load_frag2(pp, pp + 16);
      #pragma unroll
      for (int nt2 = 0; nt2 < 8; ++nt2) {
        const bf16* vp = &Vt[nt2 * 16 + lr][kc + half * 8];
        v16bf bv = load_frag2(vp, vp + 16);
        o[nt2] = wmma_bf16(ap, bv, o[nt2]);
      }
    }
    __syncthreads();
  }

  // normalize and write O (bf16)
  #pragma unroll
  for (int nt2 = 0; nt2 < 8; ++nt2)
    #pragma unroll
    for (int r = 0; r < 8; ++r) {
      long rowg = (long)b * T + q0 + wave * 16 + r + half * 8;
      int col = h * 128 + nt2 * 16 + lr;
      O[rowg * 2048 + col] = (bf16)(o[nt2][r] / lrow[r]);
    }
}

// ---------------------------------------------------------------------------
// Host launcher
// ---------------------------------------------------------------------------
extern "C" void kernel_launch(void* const* d_in, const int* in_sizes, int n_in,
                              void* d_out, int out_size, void* d_ws, size_t ws_size,
                              hipStream_t stream) {
  (void)in_sizes; (void)n_in; (void)out_size; (void)ws_size;
  const int B = 2, T = 2048, D = 2048, H = 16, HD = 128, KR = 512;
  const long M = (long)B * T;                 // 4096 rows

  const float* x  = (const float*)d_in[0];
  const float* Wd = (const float*)d_in[1];    // [512, 2048]
  const float* Wq = (const float*)d_in[2];    // [2048, 2048]
  const float* Wk = (const float*)d_in[3];    // [2048, 512]
  const float* Wv = (const float*)d_in[4];    // [2048, 512]
  const float* Wo = (const float*)d_in[5];    // [2048, 2048]
  float* out = (float*)d_out;

  // workspace carve-up (bf16 elements)
  bf16* p = (bf16*)d_ws;
  bf16* xb   = p; p += M * D;                 //  x (bf16)
  bf16* WdT  = p; p += (long)D * KR;          //  Wd^T  [2048,512]
  bf16* WqT  = p; p += (long)D * D;           //  Wq^T  [2048,2048]
  bf16* Wkb  = p; p += (long)H * HD * KR;     //  Wk    [2048,512] (no transpose)
  bf16* WvT  = p; p += (long)KR * (H * HD);   //  Wv^T  [512,2048]
  bf16* WoT  = p; p += (long)D * D;           //  Wo^T  [2048,2048]
  bf16* cb   = p; p += M * KR;                //  latent c      [4096,512]
  bf16* Qb   = p; p += M * D;                 //  Q = x Wq^T    [4096,2048]
  bf16* Qab  = p; p += M * H * KR;            //  Qabs [4096,16,512]
  bf16* Vb   = p; p += M * (H * HD);          //  V expanded    [4096,2048]
  bf16* Ob   = p; p += M * (H * HD);          //  attn out      [4096,2048]

  // 1) precision casts / weight transposes
  cast_bf16_kernel<<<1024, 256, 0, stream>>>(x, xb, M * D);
  cast_bf16_kernel<<<512, 256, 0, stream>>>(Wk, Wkb, (long)H * HD * KR);
  transpose_bf16_kernel<<<dim3(D / 32, KR / 32), 256, 0, stream>>>(Wd, WdT, KR, D);
  transpose_bf16_kernel<<<dim3(D / 32, D / 32), 256, 0, stream>>>(Wq, WqT, D, D);
  transpose_bf16_kernel<<<dim3(KR / 32, (H * HD) / 32), 256, 0, stream>>>(Wv, WvT, H * HD, KR);
  transpose_bf16_kernel<<<dim3((H * HD) / 32, D / 32), 256, 0, stream>>>(Wo, WoT, D, H * HD);

  // 2) c = x Wd^T : [4096,512]
  gemm_bf16_kernel<false><<<dim3(KR / 128, M / 64, 1), 128, 0, stream>>>(
      xb, D, 0, WdT, KR, 0, cb, KR, 0, (int)M, KR, D);

  // 3) Q = x Wq^T : [4096,2048]
  gemm_bf16_kernel<false><<<dim3(D / 128, M / 64, 1), 128, 0, stream>>>(
      xb, D, 0, WqT, D, 0, Qb, D, 0, (int)M, D, D);

  // 4) Qabs[h] = Q_h Wk_h : batched over heads -> [4096,16,512]
  gemm_bf16_kernel<false><<<dim3(KR / 128, M / 64, H), 128, 0, stream>>>(
      Qb, D, HD, Wkb, KR, (long)HD * KR, Qab, (long)H * KR, KR, (int)M, KR, HD);

  // 5) V = c Wv^T : [4096,2048]
  gemm_bf16_kernel<false><<<dim3((H * HD) / 128, M / 64, 1), 128, 0, stream>>>(
      cb, KR, 0, WvT, H * HD, 0, Vb, H * HD, 0, (int)M, H * HD, KR);

  // 6) causal flash attention -> Ob [4096,2048]
  mla_attn_kernel<<<dim3(B * H, T / ATT_QT), 128, 0, stream>>>(Qab, cb, Vb, Ob, T);

  // 7) out = Ob Wo^T : [4096,2048] f32
  gemm_bf16_kernel<true><<<dim3(D / 128, M / 64, 1), 128, 0, stream>>>(
      Ob, H * HD, 0, WoT, D, 0, out, D, 0, (int)M, D, H * HD);
}
